// AnyModalMirasol_78340203479480
// MI455X (gfx1250) — matrix operand order repaired and checked by reading, not compiled, not run on hardware
//
#include <hip/hip_runtime.h>

// ---------------------------------------------------------------------------
// Mirasol dual (local/global) causal self-attention stack for MI455X (gfx1250)
// bf16 WMMA (v_wmma_f32_16x16x32_bf16) for all GEMMs + flash-style attention,
// fp32 accumulation, wave32. Weights pre-transposed once (bf16) so the GEMM
// hot loop stages LDS with pure b128 copies; 8 waves x (64x64) tiles, BK=64.
// Split-row epilogue uses exact magic-multiply division (no v_rcp expansion).
// Workspace use: ~87 MB.
// ---------------------------------------------------------------------------

typedef __attribute__((ext_vector_type(16))) __bf16 v16bf;
typedef __attribute__((ext_vector_type(8)))  float  v8f;
typedef __attribute__((ext_vector_type(4)))  unsigned int u32x4;

union B16x16 { v16bf v; u32x4 u[2]; unsigned short s[16]; };
union U4S    { u32x4 u; unsigned short s[8]; };

static __device__ __forceinline__ unsigned short f2b(float x) {
  unsigned int u = __float_as_uint(x);
  u += 0x7FFFu + ((u >> 16) & 1u);           // round-to-nearest-even
  return (unsigned short)(u >> 16);
}
static __device__ __forceinline__ float b2f(unsigned short s) {
  return __uint_as_float(((unsigned int)s) << 16);
}
static __device__ __forceinline__ v8f v8zero() {
  v8f z;
#pragma unroll
  for (int i = 0; i < 8; i++) z[i] = 0.0f;
  return z;
}
static __device__ __forceinline__ v8f wmma_bf16(B16x16 a, B16x16 b, v8f c) {
  return __builtin_amdgcn_wmma_f32_16x16x32_bf16(false, a.v, false, b.v,
                                                 (short)0, c, false, false);
}

// ---------------------------------------------------------------------------
// Elementwise helpers
// ---------------------------------------------------------------------------
__global__ void cvt_f32_bf16_kernel(const float* __restrict__ s,
                                    unsigned short* __restrict__ d, long long n) {
  long long i = (long long)blockIdx.x * blockDim.x + threadIdx.x;
  long long st = (long long)gridDim.x * blockDim.x;
  for (; i < n; i += st) d[i] = f2b(s[i]);
}

__global__ void copy_f32_kernel(const float* __restrict__ s,
                                float* __restrict__ d, long long n) {
  long long i = (long long)blockIdx.x * blockDim.x + threadIdx.x;
  long long st = (long long)gridDim.x * blockDim.x;
  for (; i < n; i += st) d[i] = s[i];
}

// Tiled transpose + convert: Wt[n*K + k] = bf16(W[k*N + n]).
// grid = (K/32, N/32), block = (32, 8).
__global__ __launch_bounds__(256) void transpose_f32_bf16_kernel(
    const float* __restrict__ W, unsigned short* __restrict__ Wt, int K, int N) {
  __shared__ float tile[32][33];
  int k0 = blockIdx.x * 32, n0 = blockIdx.y * 32;
  int tx = threadIdx.x, ty = threadIdx.y;
#pragma unroll
  for (int j = 0; j < 32; j += 8)
    tile[ty + j][tx] = W[(size_t)(k0 + ty + j) * N + n0 + tx];
  __syncthreads();
#pragma unroll
  for (int j = 0; j < 32; j += 8)
    Wt[(size_t)(n0 + ty + j) * K + k0 + tx] = f2b(tile[tx][ty + j]);
}

// dst[b][s][c] (bf16) = s<M0 ? memp[b][s][c] : xp[b][s-M0][c]
// grid = (M0+T, nB)
__global__ __launch_bounds__(256) void pack_concat_kernel(
    const float* __restrict__ memp, const float* __restrict__ xp,
    unsigned short* __restrict__ dst, int M0, int T, int C) {
  int s = blockIdx.x, b = blockIdx.y;
  int S = M0 + T;
  const float* src = (s < M0) ? memp + ((size_t)b * M0 + s) * C
                              : xp + ((size_t)b * T + (s - M0)) * C;
  unsigned short* d = dst + ((size_t)b * S + s) * C;
  for (int c = threadIdx.x; c < C; c += 256) d[c] = f2b(src[c]);
}

// ---------------------------------------------------------------------------
// bf16 GEMM: out[M,N] = A[M,K] * Wt^T  (Wt is pre-transposed: [N,K]).
// Block tile 128(M) x 256(N), BK=64; 8 waves each compute 64x64 via 16
// v_wmma_f32_16x16x32_bf16 per K-step. mode 0: bf16 out. mode 1: f32 out,
// rows split into (mem, x) destinations via magic-multiply division.
// ---------------------------------------------------------------------------
#define BM 128
#define BN 256
#define BK 64

__global__ __launch_bounds__(256) void gemm_bf16_kernel(
    const unsigned short* __restrict__ A, const unsigned short* __restrict__ Wt,
    int M, int N, int K,
    unsigned short* __restrict__ outb,
    float* __restrict__ out_mem, float* __restrict__ out_x,
    int S, int M0, unsigned long long magic, int mode) {
  __shared__ __align__(16) unsigned short As[BM][BK + 8];  // stride 144B
  __shared__ __align__(16) unsigned short Bs[BN][BK + 8];  // Bt: [n][k]
  const int t = threadIdx.x;
  const int lane = t & 31, lm = lane & 15, lh = lane >> 4;
  const int wid = t >> 5, wm = wid & 1, wn = wid >> 1;
  const int m0 = blockIdx.y * BM, n0 = blockIdx.x * BN;

  v8f acc[4][4];
#pragma unroll
  for (int i = 0; i < 4; i++)
#pragma unroll
    for (int j = 0; j < 4; j++) acc[i][j] = v8zero();

  const int arow = t >> 1, acolh = (t & 1) * 32;
  int agr = m0 + arow; if (agr >= M) agr = M - 1;

  for (int k0 = 0; k0 < K; k0 += BK) {
    // stage A tile: 4x b128 per thread, row-major
    {
      const unsigned short* ap = A + (size_t)agr * K + k0 + acolh;
#pragma unroll
      for (int c = 0; c < 4; c++)
        *(u32x4*)&As[arow][acolh + c * 8] = *(const u32x4*)(ap + c * 8);
      __builtin_prefetch(ap + BK, 0, 1);
    }
    // stage Bt tile: one 64-K row slice per thread, 8x b128 (pre-transposed)
    {
      const unsigned short* wp = Wt + (size_t)(n0 + t) * K + k0;
#pragma unroll
      for (int c = 0; c < 8; c++)
        *(u32x4*)&Bs[t][c * 8] = *(const u32x4*)(wp + c * 8);
      __builtin_prefetch(wp + BK, 0, 1);
    }
    __syncthreads();

#pragma unroll
    for (int ks = 0; ks < 2; ks++) {
      B16x16 af[4], bf[4];
#pragma unroll
      for (int mt = 0; mt < 4; mt++) {  // A layout: K 0-7/16-23 | 8-15/24-31
        int r = wm * 64 + mt * 16 + lm;
        af[mt].u[0] = *(const u32x4*)&As[r][ks * 32 + lh * 8];
        af[mt].u[1] = *(const u32x4*)&As[r][ks * 32 + lh * 8 + 16];
      }
#pragma unroll
      for (int nt = 0; nt < 4; nt++) {  // B layout: K 0-7/16-23 | 8-15/24-31
        int n = wn * 64 + nt * 16 + lm;
        bf[nt].u[0] = *(const u32x4*)&Bs[n][ks * 32 + lh * 8];
        bf[nt].u[1] = *(const u32x4*)&Bs[n][ks * 32 + lh * 8 + 16];
      }
#pragma unroll
      for (int mt = 0; mt < 4; mt++)
#pragma unroll
        for (int nt = 0; nt < 4; nt++)
          acc[mt][nt] = wmma_bf16(af[mt], bf[nt], acc[mt][nt]);
    }
    __syncthreads();
  }

  // store (D layout: M = r + 8*(lane/16), N = lane%16)
#pragma unroll
  for (int mt = 0; mt < 4; mt++)
#pragma unroll
    for (int nt = 0; nt < 4; nt++)
#pragma unroll
      for (int r = 0; r < 8; r++) {
        int gm = m0 + wm * 64 + mt * 16 + r + 8 * lh;
        int gn = n0 + wn * 64 + nt * 16 + lm;
        if (gm < M) {
          float vv = acc[mt][nt][r];
          if (mode == 0) {
            outb[(size_t)gm * N + gn] = f2b(vv);
          } else {
            // exact for gm < 2^16, S < 2^12: magic = 2^48/S + 1
            unsigned bb = (unsigned)(((unsigned long long)(unsigned)gm * magic) >> 48);
            int ss = gm - (int)bb * S;
            if (ss < M0)
              out_mem[((size_t)bb * M0 + ss) * N + gn] = vv;
            else
              out_x[((size_t)bb * (S - M0) + (ss - M0)) * N + gn] = vv;
          }
        }
      }
}

// ---------------------------------------------------------------------------
// RMSNorm of q (which=0) or k (which=1) row in-place on bf16 qkv buffer.
// ---------------------------------------------------------------------------
__global__ __launch_bounds__(256) void rmsnorm_qk_kernel(
    unsigned short* __restrict__ qkv, const float* __restrict__ g, int Mtot) {
  int row = blockIdx.x >> 1, which = blockIdx.x & 1;
  if (row >= Mtot) return;
  unsigned short* p = qkv + (size_t)row * 3072 + which * 1024;
  int t = threadIdx.x, lane = t & 31, wid = t >> 5;
  float vals[4], ss = 0.f;
#pragma unroll
  for (int i = 0; i < 4; i++) {
    float v = b2f(p[t + i * 256]);
    vals[i] = v; ss += v * v;
  }
#pragma unroll
  for (int m = 16; m >= 1; m >>= 1) ss += __shfl_xor(ss, m, 32);
  __shared__ float red[8];
  if (lane == 0) red[wid] = ss;
  __syncthreads();
  if (t == 0) {
    float s2 = 0.f;
    for (int i = 0; i < 8; i++) s2 += red[i];
    red[0] = rsqrtf(s2 * (1.0f / 1024.0f) + 1e-6f);
  }
  __syncthreads();
  float sc = red[0];
#pragma unroll
  for (int i = 0; i < 4; i++)
    p[t + i * 256] = f2b(vals[i] * sc * g[t + i * 256]);
}

// ---------------------------------------------------------------------------
// Flash-style causal attention. One wave = (seq b, head h, 16-query tile).
// qkv: [nB*S, 3072] bf16 (q|k|v), y: [nB*S, 1024] bf16. hd=64, 16 heads.
// ---------------------------------------------------------------------------
__global__ __launch_bounds__(256) void attn_kernel(
    const unsigned short* __restrict__ qkv, unsigned short* __restrict__ y,
    int nB, int S) {
  const int NQT = S >> 4;
  const int wid = threadIdx.x >> 5, lane = threadIdx.x & 31;
  const int lm = lane & 15, lh = lane >> 4;
  long long task = (long long)blockIdx.x * 8 + wid;
  long long ntasks = (long long)nB * 16 * NQT;

  __shared__ __align__(16) unsigned short pbuf_all[8][16][40];
  __shared__ __align__(16) unsigned short vt_all[8][64][40];
  if (task >= ntasks) return;
  unsigned short (*pbuf)[40] = pbuf_all[wid];
  unsigned short (*vt)[40]   = vt_all[wid];

  int qt = (int)(task % NQT);
  int tmp = (int)(task / NQT);
  int h = tmp & 15, b = tmp >> 4;
  int qbase = qt * 16;
  size_t base = (size_t)b * S * 3072;

  // Q fragments (A layout), K-dim = head dim 64 -> two K32 steps
  B16x16 qf[2];
  {
    const unsigned short* qp = qkv + base + (size_t)(qbase + lm) * 3072 + h * 64;
#pragma unroll
    for (int ks = 0; ks < 2; ks++) {
      qf[ks].u[0] = *(const u32x4*)(qp + ks * 32 + lh * 8);
      qf[ks].u[1] = *(const u32x4*)(qp + ks * 32 + lh * 8 + 16);
    }
  }

  v8f o[4];
#pragma unroll
  for (int i = 0; i < 4; i++) o[i] = v8zero();
  float mrow[8], lrow[8];
#pragma unroll
  for (int r = 0; r < 8; r++) { mrow[r] = -3e38f; lrow[r] = 0.f; }

  const int jend = qbase + 16;
  for (int j0 = 0; j0 < jend; j0 += 32) {
    // scores S = Q*K^T : K rows are already Bt row-major
    v8f sc[2] = {v8zero(), v8zero()};
#pragma unroll
    for (int nt = 0; nt < 2; nt++) {
      int kj = j0 + nt * 16 + lm;
      if (kj >= S) kj = S - 1;                       // clamped; masked below
      const unsigned short* kp =
          qkv + base + (size_t)kj * 3072 + 1024 + h * 64;
#pragma unroll
      for (int ks = 0; ks < 2; ks++) {
        B16x16 kf;
        kf.u[0] = *(const u32x4*)(kp + ks * 32 + lh * 8);
        kf.u[1] = *(const u32x4*)(kp + ks * 32 + lh * 8 + 16);
        sc[nt] = wmma_bf16(qf[ks], kf, sc[nt]);
      }
    }

    // scale + causal mask + online softmax
    float pv[2][8], bm[8];
#pragma unroll
    for (int r = 0; r < 8; r++) {
      int qi = qbase + r + 8 * lh;
      float m2 = -3e38f;
#pragma unroll
      for (int nt = 0; nt < 2; nt++) {
        int kj = j0 + nt * 16 + lm;
        float sv = sc[nt][r] * 0.125f;               // 1/sqrt(64)
        if (kj > qi) sv = -3e38f;
        pv[nt][r] = sv;
        m2 = fmaxf(m2, sv);
      }
      bm[r] = m2;
    }
#pragma unroll
    for (int m = 1; m < 16; m <<= 1)
#pragma unroll
      for (int r = 0; r < 8; r++)
        bm[r] = fmaxf(bm[r], __shfl_xor(bm[r], m, 32));

    float corr[8], rs[8];
#pragma unroll
    for (int r = 0; r < 8; r++) {
      float nm = fmaxf(mrow[r], bm[r]);
      corr[r] = __expf(mrow[r] - nm);
      mrow[r] = nm;
      float srow = 0.f;
#pragma unroll
      for (int nt = 0; nt < 2; nt++) {
        float pe = __expf(pv[nt][r] - nm);
        pv[nt][r] = pe; srow += pe;
      }
      rs[r] = srow;
    }
#pragma unroll
    for (int m = 1; m < 16; m <<= 1)
#pragma unroll
      for (int r = 0; r < 8; r++) rs[r] += __shfl_xor(rs[r], m, 32);
#pragma unroll
    for (int r = 0; r < 8; r++) lrow[r] = lrow[r] * corr[r] + rs[r];
#pragma unroll
    for (int t4 = 0; t4 < 4; t4++)
#pragma unroll
      for (int r = 0; r < 8; r++) o[t4][r] *= corr[r];

    // P: D-layout -> LDS -> A-layout bf16 fragment
#pragma unroll
    for (int nt = 0; nt < 2; nt++)
#pragma unroll
      for (int r = 0; r < 8; r++)
        pbuf[r + 8 * lh][nt * 16 + lm] = f2b(pv[nt][r]);

    // V block transposed into LDS: vt[d][key]
    {
      int kj = j0 + lane;
      int kjc = (kj < S) ? kj : (S - 1);
      const unsigned short* vp =
          qkv + base + (size_t)kjc * 3072 + 2048 + h * 64;
#pragma unroll
      for (int c2 = 0; c2 < 8; c2++) {
        U4S vv; vv.u = *(const u32x4*)(vp + c2 * 8);
#pragma unroll
        for (int jj = 0; jj < 8; jj++) vt[c2 * 8 + jj][lane] = vv.s[jj];
      }
    }
    asm volatile("s_wait_dscnt 0x0" ::: "memory");

    B16x16 pf;
    pf.u[0] = *(const u32x4*)&pbuf[lm][lh * 8];
    pf.u[1] = *(const u32x4*)&pbuf[lm][lh * 8 + 16];
#pragma unroll
    for (int t4 = 0; t4 < 4; t4++) {
      B16x16 vf;
      vf.u[0] = *(const u32x4*)&vt[t4 * 16 + lm][lh * 8];
      vf.u[1] = *(const u32x4*)&vt[t4 * 16 + lm][lh * 8 + 16];
      o[t4] = wmma_bf16(pf, vf, o[t4]);
    }
  }

  // normalize + store y (bf16)
#pragma unroll
  for (int t4 = 0; t4 < 4; t4++)
#pragma unroll
    for (int r = 0; r < 8; r++) {
      int qrow = qbase + r + 8 * lh;
      float val = o[t4][r] / lrow[r];
      y[((size_t)b * S + qrow) * 1024 + h * 64 + t4 * 16 + lm] = f2b(val);
    }
}

// ---------------------------------------------------------------------------
// Host orchestration
// ---------------------------------------------------------------------------
static inline unsigned gsblocks(long long n) {
  long long bl = (n + 1023) / 1024;
  if (bl > 65535) bl = 65535;
  if (bl < 1) bl = 1;
  return (unsigned)bl;
}

extern "C" void kernel_launch(void* const* d_in, const int* in_sizes, int n_in,
                              void* d_out, int out_size, void* d_ws, size_t ws_size,
                              hipStream_t stream) {
  (void)in_sizes; (void)n_in; (void)out_size; (void)ws_size;
  const float* x_in    = (const float*)d_in[0];
  const float* mem_in  = (const float*)d_in[1];
  const float* lm_in   = (const float*)d_in[2];
  const float* wqkv_l  = (const float*)d_in[3];
  const float* wproj_l = (const float*)d_in[4];
  const float* g_l     = (const float*)d_in[5];
  const float* wqkv_g  = (const float*)d_in[6];
  const float* wproj_g = (const float*)d_in[7];
  const float* g_g     = (const float*)d_in[8];

  const int C = 1024, C3 = 3072;
  const int SL = 272, SG = 2064;        // local/global sequence lengths
  const int MT_L = 16 * SL;             // 4352 rows
  const int MT_G = 2 * SG;              // 4128 rows
  const long long NX = 2LL * 8 * 256 * C;
  const unsigned long long magicL = (1ULL << 48) / SL + 1;
  const unsigned long long magicG = (1ULL << 48) / SG + 1;

  char* p = (char*)d_ws;
  auto alloc = [&](size_t bytes) -> char* {
    char* r = p; p += (bytes + 255) & ~(size_t)255; return r;
  };
  float* xbuf  = (float*)alloc((size_t)NX * 4);
  float* lmbuf = (float*)alloc((size_t)2 * 8 * 16 * C * 4);
  float* mtbuf = (float*)alloc((size_t)2 * 16 * C * 4);
  unsigned short* xin  = (unsigned short*)alloc((size_t)MT_L * C * 2);
  unsigned short* ybuf = xin;  // alias: xin consumed before y produced
  unsigned short* qkv  = (unsigned short*)alloc((size_t)MT_L * C3 * 2);
  unsigned short* wql  = (unsigned short*)alloc((size_t)2 * C3 * C * 2);  // [i][N][K]
  unsigned short* wpl  = (unsigned short*)alloc((size_t)2 * C * C * 2);
  unsigned short* wqg  = (unsigned short*)alloc((size_t)2 * C3 * C * 2);
  unsigned short* wpg  = (unsigned short*)alloc((size_t)2 * C * C * 2);

  dim3 tb(32, 8);
  for (int i = 0; i < 2; i++) {
    transpose_f32_bf16_kernel<<<dim3(C / 32, C3 / 32), tb, 0, stream>>>(
        wqkv_l + (size_t)i * C * C3, wql + (size_t)i * C3 * C, C, C3);
    transpose_f32_bf16_kernel<<<dim3(C / 32, C / 32), tb, 0, stream>>>(
        wproj_l + (size_t)i * C * C, wpl + (size_t)i * C * C, C, C);
    transpose_f32_bf16_kernel<<<dim3(C / 32, C3 / 32), tb, 0, stream>>>(
        wqkv_g + (size_t)i * C * C3, wqg + (size_t)i * C3 * C, C, C3);
    transpose_f32_bf16_kernel<<<dim3(C / 32, C / 32), tb, 0, stream>>>(
        wproj_g + (size_t)i * C * C, wpg + (size_t)i * C * C, C, C);
  }
  copy_f32_kernel<<<gsblocks(NX), 256, 0, stream>>>(x_in, xbuf, NX);
  copy_f32_kernel<<<gsblocks(2LL*8*16*C), 256, 0, stream>>>(lm_in, lmbuf, 2LL*8*16*C);
  copy_f32_kernel<<<gsblocks(2LL*16*C), 256, 0, stream>>>(mem_in, mtbuf, 2LL*16*C);

  for (int i = 0; i < 2; i++) {
    // ---- local attention over 16 sequences of 272 ----
    pack_concat_kernel<<<dim3(SL, 16), 256, 0, stream>>>(lmbuf, xbuf, xin, 16, 256, C);
    {
      dim3 g(C3 / BN, (MT_L + BM - 1) / BM);
      gemm_bf16_kernel<<<g, 256, 0, stream>>>(xin, wql + (size_t)i * C3 * C,
                                              MT_L, C3, C, qkv, nullptr, nullptr,
                                              0, 0, 0ULL, 0);
    }
    rmsnorm_qk_kernel<<<MT_L * 2, 256, 0, stream>>>(qkv, g_l + (size_t)i * C, MT_L);
    {
      int ntasks = 16 * 16 * (SL / 16);
      attn_kernel<<<(ntasks + 7) / 8, 256, 0, stream>>>(qkv, ybuf, 16, SL);
    }
    {
      dim3 g(C / BN, (MT_L + BM - 1) / BM);
      gemm_bf16_kernel<<<g, 256, 0, stream>>>(ybuf, wpl + (size_t)i * C * C,
                                              MT_L, C, C, nullptr, lmbuf, xbuf,
                                              SL, 16, magicL, 1);
    }
    // ---- global attention over 2 sequences of 2064 ----
    pack_concat_kernel<<<dim3(SG, 2), 256, 0, stream>>>(mtbuf, xbuf, xin, 16, 2048, C);
    {
      dim3 g(C3 / BN, (MT_G + BM - 1) / BM);
      gemm_bf16_kernel<<<g, 256, 0, stream>>>(xin, wqg + (size_t)i * C3 * C,
                                              MT_G, C3, C, qkv, nullptr, nullptr,
                                              0, 0, 0ULL, 0);
    }
    rmsnorm_qk_kernel<<<MT_G * 2, 256, 0, stream>>>(qkv, g_g + (size_t)i * C, MT_G);
    {
      int ntasks = 2 * 16 * (SG / 16);
      attn_kernel<<<(ntasks + 7) / 8, 256, 0, stream>>>(qkv, ybuf, 2, SG);
    }
    {
      float* xdst = (i == 1) ? (float*)d_out : xbuf;   // final x goes to d_out
      dim3 g(C / BN, (MT_G + BM - 1) / BM);
      gemm_bf16_kernel<<<g, 256, 0, stream>>>(ybuf, wpg + (size_t)i * C * C,
                                              MT_G, C, C, nullptr, mtbuf, xdst,
                                              SG, 16, magicG, 1);
    }
  }
}
